// Block_8031588843661
// MI455X (gfx1250) — matrix operand order repaired and checked by reading, not compiled, not run on hardware
//
#include <hip/hip_runtime.h>
#include <math.h>

typedef __bf16 bf16_t;
typedef __attribute__((ext_vector_type(16))) __bf16 v16bf;
typedef __attribute__((ext_vector_type(8)))  float  v8f;

// ---------------------------------------------------------------------------
// WMMA fragment helpers (contiguous k order k = (l>>4)*16 + j in A and B, wave32)
// ---------------------------------------------------------------------------

// A-matrix 16x32 bf16 from row-major source (stride lda), tile K offset k0.
// lane l: row = l&15 ; K = koff + {0..15}, koff = (l>>4)*16
__device__ inline v16bf load_a(const bf16_t* src, int lda, int k0, int lane) {
    int r    = lane & 15;
    int koff = (lane >> 4) << 4;
    const bf16_t* p = src + r * lda + k0 + koff;
    v16bf a;
#pragma unroll
    for (int j = 0; j < 16; ++j) { a[j] = p[j]; }
    return a;
}

// B-matrix 32x16 bf16 where the source is stored TRANSPOSED: srcT[N][K]
// (row = output column). lane l: N = n0 + (l&15); K = k0 + (l>>4)*16 + j.
__device__ inline v16bf load_b(const bf16_t* srcT, int ldb, int n0, int k0, int lane) {
    int n  = lane & 15;
    int kb = (lane >> 4) << 4;
    const bf16_t* p = srcT + (size_t)(n0 + n) * ldb + k0 + kb;
    v16bf b;
#pragma unroll
    for (int j = 0; j < 16; ++j) b[j] = p[j];
    return b;
}

// B-matrix with K zero-padding past kvalid (used for 16-wide wei @ V).
__device__ inline v16bf load_b_pad(const bf16_t* srcT, int ldb, int n0, int kvalid, int lane) {
    int n  = lane & 15;
    int kb = (lane >> 4) << 4;
    v16bf b;
#pragma unroll
    for (int j = 0; j < 16; ++j) {
        int k = kb + j;
        b[j] = (k < kvalid) ? srcT[(size_t)(n0 + n) * ldb + k] : (bf16_t)0.0f;
    }
    return b;
}

__device__ inline v8f wmma_bf16(const v16bf& a, const v16bf& b, const v8f& c) {
    return __builtin_amdgcn_wmma_f32_16x16x32_bf16(false, a, false, b, (short)0, c,
                                                   false, false);
}

#define V8F_ZERO {0.f,0.f,0.f,0.f,0.f,0.f,0.f,0.f}

// ---------------------------------------------------------------------------
// Kernel 0: f32 -> bf16 transposed weight copy: out[n*K+k] = in[k*N+n]
// ---------------------------------------------------------------------------
__global__ void k_cvt_transpose(const float* __restrict__ in, bf16_t* __restrict__ out,
                                int K, int N) {
    int idx = blockIdx.x * blockDim.x + threadIdx.x;
    if (idx >= K * N) return;
    int n = idx / K;
    int k = idx - n * K;
    out[idx] = (bf16_t)in[(size_t)k * N + n];
}

// ---------------------------------------------------------------------------
// Kernel 1: per-b fused QKV + block-diag causal attention + proj + LN1 -> x1
// one workgroup (8 waves) per b; 16 rows (t*4+h) of 512
// ---------------------------------------------------------------------------
__global__ __launch_bounds__(256) void k_attn_ln1(
    const float* __restrict__ x,
    const bf16_t* __restrict__ WqT, const bf16_t* __restrict__ WkT,
    const bf16_t* __restrict__ WvT, const bf16_t* __restrict__ WpT,
    const float* __restrict__ bp,
    const float* __restrict__ g1, const float* __restrict__ be1,
    float* __restrict__ x1f) {
    __shared__ __align__(16) unsigned char smem[54 * 1024];
    bf16_t* Axh = (bf16_t*)(smem);             // 16x128 bf16 (4 KB)  = x[b]
    bf16_t* Qm  = (bf16_t*)(smem + 4  * 1024); // 16x512 bf16 (16 KB)
    bf16_t* Km  = (bf16_t*)(smem + 20 * 1024); // 16x512 bf16 (16 KB)
    bf16_t* Vt  = (bf16_t*)(smem + 36 * 1024); // 512x16 bf16 (16 KB) V transposed
    bf16_t* Wei = (bf16_t*)(smem + 52 * 1024); // 16x32  bf16 (1 KB), cols 16..31 zero
    bf16_t* Att = Qm;                          // alias (Q dead after scores)
    float*  Yt  = (float*)(smem + 20 * 1024);  // 16x512 f32 (32 KB), alias Km+Vt

    const int tid  = threadIdx.x;
    const int wid  = tid >> 5;
    const int lane = tid & 31;
    const int b    = blockIdx.x;
    const float* xb = x + (size_t)b * 2048;

    // stage x[b] as bf16 A-matrix rows (row (t,h) = xb + (t*4+h)*128, contiguous)
    for (int i = tid; i < 2048; i += 256) Axh[i] = (bf16_t)xb[i];
    for (int i = tid; i < 16 * 32; i += 256) Wei[i] = (bf16_t)0.0f;
    __syncthreads();

    // ---- QKV: 3 x (16x128)@(128x512) = 96 output tiles, 12 per wave ----
    for (int t = 0; t < 12; ++t) {
        int tq  = wid * 12 + t;
        int mat = tq >> 5;
        int n0  = (tq & 31) << 4;
        const bf16_t* WT = (mat == 0) ? WqT : (mat == 1) ? WkT : WvT;
        v8f acc = V8F_ZERO;
#pragma unroll
        for (int kk = 0; kk < 4; ++kk) {
            v16bf a  = load_a(Axh, 128, kk * 32, lane);
            v16bf bm = load_b(WT, 128, n0, kk * 32, lane);
            acc = wmma_bf16(a, bm, acc);
        }
        int n = lane & 15, mh = (lane >> 4) << 3;
        if (mat == 2) {               // V stored transposed: Vt[d][row]
#pragma unroll
            for (int p = 0; p < 8; ++p) Vt[(n0 + n) * 16 + (mh + p)] = (bf16_t)acc[p];
        } else {
            bf16_t* dst = (mat == 0) ? Qm : Km;
#pragma unroll
            for (int p = 0; p < 8; ++p) dst[(mh + p) * 512 + n0 + n] = (bf16_t)acc[p];
        }
    }
    __syncthreads();

    // ---- scores = Q @ K^T (16x16, K=512) + block-diagonal causal softmax ----
    if (wid == 0) {
        v8f acc = V8F_ZERO;
#pragma unroll
        for (int kk = 0; kk < 16; ++kk) {
            v16bf a  = load_a(Qm, 512, kk * 32, lane);
            v16bf bm = load_b(Km, 512, 0, kk * 32, lane);   // B[K=d][N=j]=Km[j][d]
            acc = wmma_bf16(a, bm, acc);
        }
        const float scale = 0.044194173824159216f;          // 512^-0.5
        int n = lane & 15, mh = (lane >> 4) << 3;
#pragma unroll
        for (int p = 0; p < 8; ++p) {
            int m = mh + p;                                  // row (t,i): t=m>>2, i=m&3
            bool valid = ((n >> 2) == (m >> 2)) && ((n & 3) <= (m & 3));
            float s  = valid ? acc[p] * scale : -3.0e38f;
            float mx = fmaxf(s, __shfl_xor(s, 1, 32));
            mx       = fmaxf(mx, __shfl_xor(mx, 2, 32));
            float e  = valid ? __expf(s - mx) : 0.0f;
            float sm = e;
            sm += __shfl_xor(sm, 1, 32);
            sm += __shfl_xor(sm, 2, 32);
            Wei[m * 32 + n] = (bf16_t)(valid ? e / sm : 0.0f);
        }
    }
    __syncthreads();

    // ---- att = Wei(16x32, zero-padded) @ V(32x512, rows 16..31 zero) ----
    for (int t = 0; t < 4; ++t) {
        int n0 = (wid * 4 + t) << 4;
        v8f acc = V8F_ZERO;
        v16bf a  = load_a(Wei, 32, 0, lane);
        v16bf bm = load_b_pad(Vt, 16, n0, 16, lane);
        acc = wmma_bf16(a, bm, acc);
        int n = lane & 15, mh = (lane >> 4) << 3;
#pragma unroll
        for (int p = 0; p < 8; ++p) Att[(mh + p) * 512 + n0 + n] = (bf16_t)acc[p];
    }
    __syncthreads();

    // ---- y = att @ Wp (16x512)@(512x512) ----
    for (int t = 0; t < 4; ++t) {
        int n0 = (wid * 4 + t) << 4;
        v8f acc = V8F_ZERO;
#pragma unroll
        for (int kk = 0; kk < 16; ++kk) {
            v16bf a  = load_a(Att, 512, kk * 32, lane);
            v16bf bm = load_b(WpT, 512, n0, kk * 32, lane);
            acc = wmma_bf16(a, bm, acc);
        }
        int n = lane & 15, mh = (lane >> 4) << 3;
#pragma unroll
        for (int p = 0; p < 8; ++p) Yt[(mh + p) * 512 + n0 + n] = acc[p];
    }
    __syncthreads();

    // ---- LN1 over 512 (eps = 512, faithful): x1 = LN(x + y + bp) ----
    {
        int r = tid >> 4, cg = tid & 15;                    // 16 threads / row
        const float* xr = xb + (r >> 2) * 512;              // residual broadcast over head
        float vb[32], s = 0.f, s2 = 0.f;
#pragma unroll
        for (int j = 0; j < 32; ++j) {
            int c = cg * 32 + j;
            float v = Yt[r * 512 + c] + bp[c] + xr[c];
            vb[j] = v; s += v; s2 += v * v;
        }
#pragma unroll
        for (int o = 1; o < 16; o <<= 1) {
            s += __shfl_xor(s, o, 32); s2 += __shfl_xor(s2, o, 32);
        }
        float mean = s * (1.0f / 512.0f);
        float var  = s2 * (1.0f / 512.0f) - mean * mean;
        float rstd = rsqrtf(var + 512.0f);                  // eps = head_size
        float* orow = x1f + ((size_t)b * 16 + r) * 512;
#pragma unroll
        for (int j = 0; j < 32; ++j) {
            int c = cg * 32 + j;
            orow[c] = (vb[j] - mean) * rstd * g1[c] + be1[c];
        }
    }
}

// ---------------------------------------------------------------------------
// Kernel 2: fused FFN + LN2. One workgroup per 32-row tile of x1 (M=32
// doubles B-fragment reuse vs M=16, halving W1T/W2T L2 traffic). Hidden dim
// (2048) streamed in 4 chunks of 512; 2x4 output accumulator tiles in VGPRs.
// Next chunk's weight rows are prefetched (global_prefetch_b8) during compute.
// ---------------------------------------------------------------------------
__global__ __launch_bounds__(256) void k_ffn_ln2(
    const float* __restrict__ x1f,
    const bf16_t* __restrict__ W1T, const bf16_t* __restrict__ W2T,
    const float* __restrict__ b1, const float* __restrict__ b2,
    const float* __restrict__ g2, const float* __restrict__ be2,
    float* __restrict__ out) {
    __shared__ __align__(16) unsigned char smem[64 * 1024];
    bf16_t* X1 = (bf16_t*)smem;                 // 32x512 bf16 (32 KB)
    bf16_t* Hb = (bf16_t*)(smem + 32 * 1024);   // 32x512 bf16 (32 KB)
    float*  Ob = (float*)smem;                  // 32x512 f32 (64 KB), alias

    const int tid  = threadIdx.x;
    const int wid  = tid >> 5;
    const int lane = tid & 31;
    const size_t row0 = (size_t)blockIdx.x * 32;
    const float* xs = x1f + row0 * 512;

    for (int i = tid; i < 32 * 512; i += 256) X1[i] = (bf16_t)xs[i];

    v8f oacc[2][4];
#pragma unroll
    for (int mt = 0; mt < 2; ++mt)
#pragma unroll
        for (int t = 0; t < 4; ++t) { v8f z = V8F_ZERO; oacc[mt][t] = z; }
    __syncthreads();

    for (int kb = 0; kb < 4; ++kb) {
        // speculative prefetch of next chunk's weight rows used by this wave
        if (kb < 3) {
            size_t r0 = (size_t)((kb + 1) * 512 + wid * 64 + lane);
            __builtin_prefetch(W1T + r0 * 512, 0, 1);
            __builtin_prefetch(W1T + (r0 + 32) * 512, 0, 1);
            size_t c0 = (size_t)(wid * 64 + lane);
            __builtin_prefetch(W2T + c0 * 2048 + (kb + 1) * 512, 0, 1);
            __builtin_prefetch(W2T + (c0 + 32) * 2048 + (kb + 1) * 512, 0, 1);
        }
        // h_chunk = gelu(x1 @ W1[:, kb*512 : kb*512+512] + b1)  (32x512)
        for (int t = 0; t < 4; ++t) {
            int nloc = (wid * 4 + t) << 4;       // 0..511 within chunk
            int nh   = kb * 512 + nloc;          // global hidden column
            v8f hacc0 = V8F_ZERO, hacc1 = V8F_ZERO;
#pragma unroll
            for (int kk = 0; kk < 16; ++kk) {
                v16bf bm = load_b(W1T, 512, nh, kk * 32, lane);   // shared by both M
                v16bf a0 = load_a(X1, 512, kk * 32, lane);
                hacc0 = wmma_bf16(a0, bm, hacc0);
                v16bf a1 = load_a(X1 + 16 * 512, 512, kk * 32, lane);
                hacc1 = wmma_bf16(a1, bm, hacc1);
            }
            int n = lane & 15, mh = (lane >> 4) << 3;
#pragma unroll
            for (int p = 0; p < 8; ++p) {
                float v0 = hacc0[p] + b1[nh + n];
                v0 = 0.5f * v0 * (1.0f + erff(v0 * 0.70710678118654752f));
                Hb[(mh + p) * 512 + nloc + n] = (bf16_t)v0;
                float v1 = hacc1[p] + b1[nh + n];
                v1 = 0.5f * v1 * (1.0f + erff(v1 * 0.70710678118654752f));
                Hb[(16 + mh + p) * 512 + nloc + n] = (bf16_t)v1;
            }
        }
        __syncthreads();
        // out_acc += h_chunk @ W2[kb*512 : kb*512+512, :]
        for (int t = 0; t < 4; ++t) {
            int n0 = (wid * 4 + t) << 4;
            v8f a0c = oacc[0][t], a1c = oacc[1][t];
#pragma unroll
            for (int kk = 0; kk < 16; ++kk) {
                v16bf bm = load_b(W2T, 2048, n0, kb * 512 + kk * 32, lane);
                v16bf a0 = load_a(Hb, 512, kk * 32, lane);
                a0c = wmma_bf16(a0, bm, a0c);
                v16bf a1 = load_a(Hb + 16 * 512, 512, kk * 32, lane);
                a1c = wmma_bf16(a1, bm, a1c);
            }
            oacc[0][t] = a0c; oacc[1][t] = a1c;
        }
        __syncthreads();
    }

    // bias + residual (f32 x1 from global; tile is L2-resident) into Ob
    {
        int n = lane & 15, mh = (lane >> 4) << 3;
        for (int mt = 0; mt < 2; ++mt) {
            for (int t = 0; t < 4; ++t) {
                int n0 = (wid * 4 + t) << 4;
#pragma unroll
                for (int p = 0; p < 8; ++p) {
                    int m = mt * 16 + mh + p, c = n0 + n;
                    Ob[m * 512 + c] = oacc[mt][t][p] + b2[c] + xs[(size_t)m * 512 + c];
                }
            }
        }
    }
    __syncthreads();

    // LN2 (eps = 512) -> d_out ; 32 rows handled in two passes of 16
    for (int half = 0; half < 2; ++half) {
        int r = (tid >> 4) + half * 16, cg = tid & 15;
        float vb[32], s = 0.f, s2 = 0.f;
#pragma unroll
        for (int j = 0; j < 32; ++j) {
            int c = cg * 32 + j;
            float v = Ob[r * 512 + c];
            vb[j] = v; s += v; s2 += v * v;
        }
#pragma unroll
        for (int o = 1; o < 16; o <<= 1) {
            s += __shfl_xor(s, o, 32); s2 += __shfl_xor(s2, o, 32);
        }
        float mean = s * (1.0f / 512.0f);
        float var  = s2 * (1.0f / 512.0f) - mean * mean;
        float rstd = rsqrtf(var + 512.0f);
        float* orow = out + (row0 + r) * 512;
#pragma unroll
        for (int j = 0; j < 32; ++j) {
            int c = cg * 32 + j;
            orow[c] = (vb[j] - mean) * rstd * g2[c] + be2[c];
        }
    }
}

// ---------------------------------------------------------------------------
// Host launcher
// ---------------------------------------------------------------------------
extern "C" void kernel_launch(void* const* d_in, const int* in_sizes, int n_in,
                              void* d_out, int out_size, void* d_ws, size_t ws_size,
                              hipStream_t stream) {
    (void)in_sizes; (void)n_in; (void)out_size; (void)ws_size;
    const float* x     = (const float*)d_in[0];
    const float* Wq    = (const float*)d_in[1];
    const float* Wk    = (const float*)d_in[2];
    const float* Wv    = (const float*)d_in[3];
    const float* Wp    = (const float*)d_in[4];
    const float* bp    = (const float*)d_in[5];
    const float* W1    = (const float*)d_in[6];
    const float* b1    = (const float*)d_in[7];
    const float* W2    = (const float*)d_in[8];
    const float* b2    = (const float*)d_in[9];
    const float* ln1_g = (const float*)d_in[10];
    const float* ln1_b = (const float*)d_in[11];
    const float* ln2_g = (const float*)d_in[12];
    const float* ln2_b = (const float*)d_in[13];

    unsigned char* ws = (unsigned char*)d_ws;
    bf16_t* WqT = (bf16_t*)ws;                  // [512][128]
    bf16_t* WkT = WqT + 512 * 128;
    bf16_t* WvT = WkT + 512 * 128;
    bf16_t* WpT = WvT + 512 * 128;              // [512][512]
    bf16_t* W1T = WpT + 512 * 512;              // [2048][512]
    bf16_t* W2T = W1T + 2048 * 512;             // [512][2048]
    float*  x1f = (float*)(ws + (8ull << 20));  // [65536][512] f32 (128 MB)

    k_cvt_transpose<<<(128 * 512 + 255) / 256, 256, 0, stream>>>(Wq, WqT, 128, 512);
    k_cvt_transpose<<<(128 * 512 + 255) / 256, 256, 0, stream>>>(Wk, WkT, 128, 512);
    k_cvt_transpose<<<(128 * 512 + 255) / 256, 256, 0, stream>>>(Wv, WvT, 128, 512);
    k_cvt_transpose<<<(512 * 512 + 255) / 256, 256, 0, stream>>>(Wp, WpT, 512, 512);
    k_cvt_transpose<<<(512 * 2048 + 255) / 256, 256, 0, stream>>>(W1, W1T, 512, 2048);
    k_cvt_transpose<<<(2048 * 512 + 255) / 256, 256, 0, stream>>>(W2, W2T, 2048, 512);

    k_attn_ln1<<<4096, 256, 0, stream>>>(x, WqT, WkT, WvT, WpT, bp, ln1_g, ln1_b, x1f);
    k_ffn_ln2 <<<2048, 256, 0, stream>>>(x1f, W1T, W2T, b1, b2, ln2_g, ln2_b,
                                         (float*)d_out);
}
